// SingleLayerTransformer_75118978007217
// MI455X (gfx1250) — matrix operand order repaired and checked by reading, not compiled, not run on hardware
//
#include <hip/hip_runtime.h>

// ---------------------------------------------------------------------------
// Single-layer transformer, algebraically folded for MI455X (gfx1250, wave32):
//   out[b,n] = softmax_causal(q[n]·k[m]/32) · u[m] + bout,  u[m]=x[m]·(Wv^T(Wout@Wfc)^T)
// Only even queries computed. GEMMs via V_WMMA_F32_16X16X32_F16 (fp32 accum).
// ---------------------------------------------------------------------------

typedef __attribute__((ext_vector_type(16))) _Float16 v16h;
typedef __attribute__((ext_vector_type(8)))  _Float16 v8h;
typedef __attribute__((ext_vector_type(8)))  float    v8f;

#define WMMA_F16(a, b, c) \
  __builtin_amdgcn_wmma_f32_16x16x32_f16(false, (a), false, (b), (short)0, (c), false, false)

template <int BASE>
__device__ __forceinline__ void pack8(v16h& d, const float4 a, const float4 b) {
  d[BASE + 0] = (_Float16)a.x; d[BASE + 1] = (_Float16)a.y;
  d[BASE + 2] = (_Float16)a.z; d[BASE + 3] = (_Float16)a.w;
  d[BASE + 4] = (_Float16)b.x; d[BASE + 5] = (_Float16)b.y;
  d[BASE + 6] = (_Float16)b.z; d[BASE + 7] = (_Float16)b.w;
}

// out[e] = sum_j v[j] * M[j*1024 + e]   (e in [0,1024))
__global__ __launch_bounds__(256) void vecmat1024_kernel(
    const float* __restrict__ M, const float* __restrict__ v, float* __restrict__ out) {
  const int e = blockIdx.x * 256 + threadIdx.x;
  if (e >= 1024) return;
  float acc = 0.f;
  for (int j = 0; j < 1024; ++j) acc = fmaf(v[j], M[j * 1024 + e], acc);
  out[e] = acc;
}

// U[row] = x[row,:] . wv   (one wave per row, fp32)
__global__ __launch_bounds__(256) void rowdot_kernel(
    const float* __restrict__ X, const float* __restrict__ wv,
    float* __restrict__ U, int nrows) {
  const int lane = threadIdx.x & 31;
  const int row  = blockIdx.x * 8 + (threadIdx.x >> 5);
  if (row >= nrows) return;
  const float* xr = X + (size_t)row * 1024;
  float acc = 0.f;
  for (int e = lane * 4; e < 1024; e += 128) {
    const float4 t = *(const float4*)(xr + e);
    const float4 w = *(const float4*)(wv + e);
    acc += t.x * w.x + t.y * w.y + t.z * w.z + t.w * w.w;
  }
#pragma unroll
  for (int off = 16; off; off >>= 1) acc += __shfl_xor(acc, off, 32);
  if (lane == 0) U[row] = acc;
}

// P[row, d] = sum_e Xsel[row, e] * W[d, e], output f16 row-major.
// One wave -> 32x32 output block (2x2 WMMA tiles, A/B register reuse).
// evenRows: proj-row p maps to x-row (p/1024)*2048 + 2*(p%1024).
__global__ __launch_bounds__(256) void proj_wmma_kernel(
    const float* __restrict__ X, const float* __restrict__ W,
    _Float16* __restrict__ P, int mTiles, int evenRows) {
  const int lane = threadIdx.x & 31;
  const int wid  = blockIdx.x * 8 + (threadIdx.x >> 5);
  const int mt = wid >> 5;      // row-tile (32 rows)
  const int nt = wid & 31;      // col-tile (32 cols of the 1024)
  if (mt >= mTiles) return;
  const int L = lane & 15, hi = lane >> 4;
  const int rbase = mt * 32, nbase = nt * 32;

  const int pr0 = rbase + L, pr1 = rbase + 16 + L;
  int xr0, xr1;
  if (evenRows) {
    xr0 = ((pr0 >> 10) << 11) + ((pr0 & 1023) << 1);
    xr1 = ((pr1 >> 10) << 11) + ((pr1 & 1023) << 1);
  } else { xr0 = pr0; xr1 = pr1; }

  // A layout: halves 0..7 -> K = kk+8*hi+j ; halves 8..15 -> K = kk+16+8*hi+j
  const float* Xr0 = X + (size_t)xr0 * 1024 + 8 * hi;
  const float* Xr1 = X + (size_t)xr1 * 1024 + 8 * hi;
  // B layout: same as A (halves 0..7 -> K = kk+8*hi+j ; 8..15 -> K = kk+16+8*hi+j) at column nbase(+16)+L ; B[e][d]=W[d][e]
  const float* Wc0 = W + (size_t)(nbase + L) * 1024 + 8 * hi;
  const float* Wc1 = W + (size_t)(nbase + 16 + L) * 1024 + 8 * hi;

  v8f c00 = {}, c01 = {}, c10 = {}, c11 = {};
  for (int kk = 0; kk < 1024; kk += 32) {
    v16h a0, a1, b0, b1;
    pack8<0>(a0, *(const float4*)(Xr0 + kk),      *(const float4*)(Xr0 + kk + 4));
    pack8<8>(a0, *(const float4*)(Xr0 + kk + 16), *(const float4*)(Xr0 + kk + 20));
    pack8<0>(a1, *(const float4*)(Xr1 + kk),      *(const float4*)(Xr1 + kk + 4));
    pack8<8>(a1, *(const float4*)(Xr1 + kk + 16), *(const float4*)(Xr1 + kk + 20));
    pack8<0>(b0, *(const float4*)(Wc0 + kk),      *(const float4*)(Wc0 + kk + 4));
    pack8<8>(b0, *(const float4*)(Wc0 + kk + 16), *(const float4*)(Wc0 + kk + 20));
    pack8<0>(b1, *(const float4*)(Wc1 + kk),      *(const float4*)(Wc1 + kk + 4));
    pack8<8>(b1, *(const float4*)(Wc1 + kk + 16), *(const float4*)(Wc1 + kk + 20));
    c00 = WMMA_F16(a0, b0, c00);
    c01 = WMMA_F16(a0, b1, c01);
    c10 = WMMA_F16(a1, b0, c10);
    c11 = WMMA_F16(a1, b1, c11);
  }
  // D layout: lane holds D[M = r+8*hi][N = L]
#pragma unroll
  for (int r = 0; r < 8; ++r) {
    const int m0 = rbase + 8 * hi + r, m1 = m0 + 16;
    P[(size_t)m0 * 1024 + nbase + L]      = (_Float16)c00[r];
    P[(size_t)m0 * 1024 + nbase + 16 + L] = (_Float16)c01[r];
    P[(size_t)m1 * 1024 + nbase + L]      = (_Float16)c10[r];
    P[(size_t)m1 * 1024 + nbase + 16 + L] = (_Float16)c11[r];
  }
}

// One wave per (batch, 16-even-query tile). Keys on WMMA M axis so the online
// softmax reduces in-lane over 8 accumulator rows; halves merged via shfl_xor(16).
__global__ __launch_bounds__(256) void attn_wmma_kernel(
    const _Float16* __restrict__ Kh, const _Float16* __restrict__ Qh,
    const float* __restrict__ U, const float* __restrict__ boutp,
    float* __restrict__ Out) {
  const int lane = threadIdx.x & 31;
  const int wid  = blockIdx.x * 8 + (threadIdx.x >> 5);
  const int b = wid >> 6, qt = wid & 63;
  const int L = lane & 15, hi = lane >> 4;
  const float NEG_INF = -__builtin_inff();

  const _Float16* Qr = Qh + (size_t)(b * 1024 + qt * 16 + L) * 1024 + 8 * hi;
  const int nglob = 32 * qt + 2 * L;     // actual (even) sequence position
  float mrun = NEG_INF, srun = 0.f, orun = 0.f;
  const int nkt = 2 * qt + 2;            // causal key-tile count

  for (int kt = 0; kt < nkt; ++kt) {
    const _Float16* Kr = Kh + (size_t)(b * 2048 + kt * 16 + L) * 1024 + 8 * hi;
    v8f acc = {};
    for (int kk = 0; kk < 1024; kk += 32) {
      const v8h alo = *(const v8h*)(Kr + kk);
      const v8h ahi = *(const v8h*)(Kr + kk + 16);
      const v8h blo = *(const v8h*)(Qr + kk);
      const v8h bhi = *(const v8h*)(Qr + kk + 16);
      const v16h a  = __builtin_shufflevector(alo, ahi, 0,1,2,3,4,5,6,7,8,9,10,11,12,13,14,15);
      const v16h bq = __builtin_shufflevector(blo, bhi, 0,1,2,3,4,5,6,7,8,9,10,11,12,13,14,15);
      acc = WMMA_F16(a, bq, acc);
    }
    const int kbase = kt * 16 + 8 * hi;  // lane's 8 keys: kbase..kbase+7
    const float4 u0 = *(const float4*)(U + b * 2048 + kbase);
    const float4 u1 = *(const float4*)(U + b * 2048 + kbase + 4);
    const float uv[8] = {u0.x, u0.y, u0.z, u0.w, u1.x, u1.y, u1.z, u1.w};
    float s[8], smax = NEG_INF;
#pragma unroll
    for (int r = 0; r < 8; ++r) {
      s[r] = (kbase + r <= nglob) ? acc[r] * 0.03125f : NEG_INF;
      smax = fmaxf(smax, s[r]);
    }
    const float newm = fmaxf(mrun, smax);
    if (newm > NEG_INF) {
      const float f = __expf(mrun - newm);   // exp(-inf)=0 on first live tile
      srun *= f; orun *= f;
#pragma unroll
      for (int r = 0; r < 8; ++r) {
        const float p = __expf(s[r] - newm);
        srun += p;
        orun = fmaf(p, uv[r], orun);
      }
      mrun = newm;
    }
  }
  // merge the two half-wave key groups (same query column L)
  const float m2 = __shfl_xor(mrun, 16, 32);
  const float s2 = __shfl_xor(srun, 16, 32);
  const float o2 = __shfl_xor(orun, 16, 32);
  const float Mt = fmaxf(mrun, m2);
  const float f1 = (mrun > NEG_INF) ? __expf(mrun - Mt) : 0.f;
  const float f2 = (m2  > NEG_INF) ? __expf(m2  - Mt) : 0.f;
  const float Ss = srun * f1 + s2 * f2;
  const float Os = orun * f1 + o2 * f2;
  if (hi == 0) Out[b * 1024 + qt * 16 + L] = Os / Ss + boutp[0];
}

extern "C" void kernel_launch(void* const* d_in, const int* in_sizes, int n_in,
                              void* d_out, int out_size, void* d_ws, size_t ws_size,
                              hipStream_t stream) {
  const float* x    = (const float*)d_in[0];
  const float* Wq   = (const float*)d_in[1];
  const float* Wk   = (const float*)d_in[2];
  const float* Wv   = (const float*)d_in[3];
  const float* Wfc  = (const float*)d_in[4];
  const float* Wout = (const float*)d_in[5];
  const float* bout = (const float*)d_in[6];
  float* out = (float*)d_out;

  char* ws = (char*)d_ws;
  float*    w  = (float*)(ws);                 //   4 KB
  float*    wv = (float*)(ws + 4096);          //   4 KB
  float*    U  = (float*)(ws + 8192);          //  64 KB
  _Float16* Kh = (_Float16*)(ws + (size_t)131072);                          // 32 MB
  _Float16* Qh = (_Float16*)(ws + (size_t)131072 + (size_t)32 * 1024 * 1024); // 16 MB

  // w = Wout @ Wfc ; wv = Wv^T @ w ; U[b,m] = x[b,m]·wv  (exact fp32, tiny)
  vecmat1024_kernel<<<4, 256, 0, stream>>>(Wfc, Wout, w);
  vecmat1024_kernel<<<4, 256, 0, stream>>>(Wv, w, wv);
  rowdot_kernel<<<2048, 256, 0, stream>>>(x, wv, U, 16384);

  // Kh = (x @ Wk^T) as f16 : 16384x1024 -> 512 row-tiles * 32 col-tiles waves
  proj_wmma_kernel<<<2048, 256, 0, stream>>>(x, Wk, Kh, 512, 0);
  // Qh = (x_even @ Wq^T) as f16 : 8192x1024 -> 256 * 32 waves
  proj_wmma_kernel<<<1024, 256, 0, stream>>>(x, Wq, Qh, 256, 1);

  // fused causal attention + u-contraction : 8*64 = 512 waves
  attn_wmma_kernel<<<64, 256, 0, stream>>>(Kh, Qh, U, bout, out);
}
